// Net_5901285064811
// MI455X (gfx1250) — matrix operand order repaired and checked by reading, not compiled, not run on hardware
//
#include <hip/hip_runtime.h>
#include <hip/hip_bf16.h>

typedef __attribute__((ext_vector_type(16))) __bf16 v16bf;
typedef __attribute__((ext_vector_type(8)))  __bf16 v8bf;
typedef __attribute__((ext_vector_type(8)))  float  v8f;

#define B_BATCH 16
#define NNODES  2048
#define CDIM    32
#define KC      512   // K-chunk of the B panel kept in LDS (32*512*2 = 32 KB)

__device__ __forceinline__ float elu_f(float x) {
  return x > 0.f ? x : (__expf(x) - 1.f);
}

// outT[b][n][m] = bf16( sum_k X[b][m][k] * W[k][n] ),  n<32, m<2048, Kdim in {128,32}
__global__ __launch_bounds__(256) void k_transform(
    const float* __restrict__ X, const float* __restrict__ W,
    __bf16* __restrict__ outT, int Kdim)
{
  __shared__ alignas(32) __bf16 wT[CDIM * 128];   // W transposed, bf16: [n][k]
  const int tid = threadIdx.x;
  const int b   = blockIdx.y;
  for (int idx = tid; idx < CDIM * Kdim; idx += 256) {
    int n = idx / Kdim, k = idx - n * Kdim;
    wT[idx] = (__bf16)W[k * CDIM + n];
  }
  __syncthreads();

  const int lane = tid & 31, wave = tid >> 5;
  const int lo = lane & 15, hi = lane >> 4;
  const int m0 = (blockIdx.x * 8 + wave) * 16;
  const float* xrow = X + ((size_t)b * NNODES + (size_t)(m0 + lo)) * Kdim;

  v8f c0 = {}, c1 = {};
  for (int kk = 0; kk < Kdim; kk += 32) {
    // A-matrix 16x32 bf16 layout: lanes 0-15 hold K {0..7,16..23}, lanes 16-31 {8..15,24..31}
    const float* p = xrow + kk + hi * 8;
    float4 f0 = *(const float4*)(p);
    float4 f1 = *(const float4*)(p + 4);
    float4 f2 = *(const float4*)(p + 16);
    float4 f3 = *(const float4*)(p + 20);
    v16bf af;
    af[0]=(__bf16)f0.x; af[1]=(__bf16)f0.y; af[2]=(__bf16)f0.z; af[3]=(__bf16)f0.w;
    af[4]=(__bf16)f1.x; af[5]=(__bf16)f1.y; af[6]=(__bf16)f1.z; af[7]=(__bf16)f1.w;
    af[8]=(__bf16)f2.x; af[9]=(__bf16)f2.y; af[10]=(__bf16)f2.z; af[11]=(__bf16)f2.w;
    af[12]=(__bf16)f3.x; af[13]=(__bf16)f3.y; af[14]=(__bf16)f3.z; af[15]=(__bf16)f3.w;
    const int kl = kk + hi * 8;   // B layout: lanes 0-15 K {0..7,16..23}, lanes 16-31 K {8..15,24..31}
    v16bf bf0  = __builtin_shufflevector(*(const v8bf*)&wT[(size_t)lo * Kdim + kl], *(const v8bf*)&wT[(size_t)lo * Kdim + kl + 16], 0,1,2,3,4,5,6,7,8,9,10,11,12,13,14,15);
    v16bf bf1v = __builtin_shufflevector(*(const v8bf*)&wT[(size_t)(lo + 16) * Kdim + kl], *(const v8bf*)&wT[(size_t)(lo + 16) * Kdim + kl + 16], 0,1,2,3,4,5,6,7,8,9,10,11,12,13,14,15);
    c0 = __builtin_amdgcn_wmma_f32_16x16x32_bf16(false, af, false, bf0,  (short)0, c0, false, false);
    c1 = __builtin_amdgcn_wmma_f32_16x16x32_bf16(false, af, false, bf1v, (short)0, c1, false, false);
  }
  // C layout: VGPR r, lanes 0-15 -> (M=r, N=lane); lanes 16-31 -> (M=r+8, N=lane-16)
  v8bf o0, o1;
  for (int r = 0; r < 8; ++r) { o0[r] = (__bf16)c0[r]; o1[r] = (__bf16)c1[r]; }
  __bf16* obase = outT + (size_t)b * CDIM * NNODES + m0 + hi * 8;
  *(v8bf*)(obase + (size_t)lo * NNODES)        = o0;
  *(v8bf*)(obase + (size_t)(lo + 16) * NNODES) = o1;
}

// mode 0: h_out[b][m][c] = elu(A@XW + bias), optionally caching bf16(A) to Abf
// mode 1: pooled[b][c]  += sum_m elu(A@XW + bias), optionally reading bf16(A) from Abf
__global__ __launch_bounds__(256) void k_aggregate(
    const float*  __restrict__ A,      // [B][N][N] fp32
    __bf16*       __restrict__ Abf,    // [B][N][N] bf16 cache (workspace)
    const __bf16* __restrict__ hT,     // [B][CDIM][N] bf16 (transposed X@W)
    const float*  __restrict__ bias,   // [CDIM]
    float*        __restrict__ h_out,  // [B][N][CDIM]  (mode 0)
    float*        __restrict__ pooled, // [B][CDIM]     (mode 1)
    int mode, int use_abf)
{
  __shared__ alignas(32) __bf16 bT[CDIM * KC];
  const int tid = threadIdx.x;
  const int b   = blockIdx.y;
  const int lane = tid & 31, wave = tid >> 5;
  const int lo = lane & 15, hi = lane >> 4;
  const int m0 = (blockIdx.x * 8 + wave) * 16;
  const size_t rowoff = ((size_t)b * NNODES + (size_t)(m0 + lo)) * NNODES;
  const float*  arow  = A + rowoff;
  __bf16*       abrow = Abf + rowoff;
  const __bf16* hTb   = hT + (size_t)b * CDIM * NNODES;

  v8f c0 = {}, c1 = {};
  for (int kb = 0; kb < NNODES; kb += KC) {
    __syncthreads();
    // cooperatively stage B panel chunk: bT[n][k] = hT[b][n][kb+k], 32 KB
    for (int i = 0; i < (CDIM * KC) / (256 * 8); ++i) {
      int idx = tid + i * 256;
      int n = idx >> 6, j = idx & 63;   // 64 float4 per 512-element row
      ((float4*)bT)[idx] = ((const float4*)(hTb + (size_t)n * NNODES + kb))[j];
    }
    __syncthreads();
    if (kb + KC < NNODES) {  // prefetch next A chunk of this row (global_prefetch_b8)
      if (mode == 1 && use_abf) __builtin_prefetch(abrow + kb + KC + lane * 16, 0, 0);
      else                      __builtin_prefetch(arow  + kb + KC + lane * 16, 0, 0);
    }
    for (int kk = 0; kk < KC; kk += 32) {
      const int kg = kb + kk;
      v16bf af;
      if (mode == 1 && use_abf) {
        v8bf r0 = *(const v8bf*)(abrow + kg + hi * 8);
        v8bf r1 = *(const v8bf*)(abrow + kg + 16 + hi * 8);
        af = __builtin_shufflevector(r0, r1, 0,1,2,3,4,5,6,7,8,9,10,11,12,13,14,15);
      } else {
        const float* p = arow + kg + hi * 8;
        float4 f0 = *(const float4*)(p);
        float4 f1 = *(const float4*)(p + 4);
        float4 f2 = *(const float4*)(p + 16);
        float4 f3 = *(const float4*)(p + 20);
        af[0]=(__bf16)f0.x; af[1]=(__bf16)f0.y; af[2]=(__bf16)f0.z; af[3]=(__bf16)f0.w;
        af[4]=(__bf16)f1.x; af[5]=(__bf16)f1.y; af[6]=(__bf16)f1.z; af[7]=(__bf16)f1.w;
        af[8]=(__bf16)f2.x; af[9]=(__bf16)f2.y; af[10]=(__bf16)f2.z; af[11]=(__bf16)f2.w;
        af[12]=(__bf16)f3.x; af[13]=(__bf16)f3.y; af[14]=(__bf16)f3.z; af[15]=(__bf16)f3.w;
        if (mode == 0 && use_abf) {  // cache bf16 A (L2-resident for second pass)
          *(v8bf*)(abrow + kg + hi * 8)      = __builtin_shufflevector(af, af, 0,1,2,3,4,5,6,7);
          *(v8bf*)(abrow + kg + 16 + hi * 8) = __builtin_shufflevector(af, af, 8,9,10,11,12,13,14,15);
        }
      }
      const int kl = kk + hi * 8;
      v16bf bf0  = __builtin_shufflevector(*(const v8bf*)&bT[(size_t)lo * KC + kl], *(const v8bf*)&bT[(size_t)lo * KC + kl + 16], 0,1,2,3,4,5,6,7,8,9,10,11,12,13,14,15);
      v16bf bf1v = __builtin_shufflevector(*(const v8bf*)&bT[(size_t)(lo + 16) * KC + kl], *(const v8bf*)&bT[(size_t)(lo + 16) * KC + kl + 16], 0,1,2,3,4,5,6,7,8,9,10,11,12,13,14,15);
      c0 = __builtin_amdgcn_wmma_f32_16x16x32_bf16(false, af, false, bf0,  (short)0, c0, false, false);
      c1 = __builtin_amdgcn_wmma_f32_16x16x32_bf16(false, af, false, bf1v, (short)0, c1, false, false);
    }
  }

  const int n0 = lo, n1 = lo + 16;
  const float bias0 = bias[n0], bias1 = bias[n1];
  if (mode == 0) {
    float* orow = h_out + ((size_t)b * NNODES + m0 + hi * 8) * CDIM;
    for (int r = 0; r < 8; ++r) {
      orow[(size_t)r * CDIM + n0] = elu_f(c0[r] + bias0);
      orow[(size_t)r * CDIM + n1] = elu_f(c1[r] + bias1);
    }
  } else {
    float s0 = 0.f, s1 = 0.f;
    for (int r = 0; r < 8; ++r) { s0 += elu_f(c0[r] + bias0); s1 += elu_f(c1[r] + bias1); }
    s0 += __shfl_xor(s0, 16, 32);   // combine M halves (lanes l and l+16 share column N)
    s1 += __shfl_xor(s1, 16, 32);
    if (lane < 16) {
      atomicAdd(&pooled[b * CDIM + n0], s0);
      atomicAdd(&pooled[b * CDIM + n1], s1);
    }
  }
}

__global__ void k_zero(float* __restrict__ p, int n) {
  int i = blockIdx.x * blockDim.x + threadIdx.x;
  if (i < n) p[i] = 0.f;
}

// pooled[16][32] -> relu(@wf1[32][512]+bf1) -> sigmoid(@wf2[512]+bf2) -> out[16]
__global__ __launch_bounds__(512) void k_mlp(
    const float* __restrict__ pooled, const float* __restrict__ wf1,
    const float* __restrict__ bf1,    const float* __restrict__ wf2,
    const float* __restrict__ bf2,    float* __restrict__ out)
{
  __shared__ float hid[B_BATCH * 512];
  const int t = threadIdx.x;
  for (int b = 0; b < B_BATCH; ++b) {
    float acc = bf1[t];
    for (int c = 0; c < CDIM; ++c) acc += pooled[b * CDIM + c] * wf1[c * 512 + t];
    hid[b * 512 + t] = fmaxf(acc, 0.f);
  }
  __syncthreads();
  const int wave = t >> 5, lane = t & 31;   // 16 waves -> 16 batches
  float part = 0.f;
  for (int j = lane; j < 512; j += 32) part += hid[wave * 512 + j] * wf2[j];
  for (int off = 16; off > 0; off >>= 1) part += __shfl_down(part, off, 32);
  if (lane == 0) out[wave] = 1.f / (1.f + __expf(-(part + bf2[0])));
}

extern "C" void kernel_launch(void* const* d_in, const int* in_sizes, int n_in,
                              void* d_out, int out_size, void* d_ws, size_t ws_size,
                              hipStream_t stream) {
  const float* x   = (const float*)d_in[0];
  const float* a   = (const float*)d_in[1];
  const float* w1  = (const float*)d_in[2];
  const float* b1  = (const float*)d_in[3];
  const float* w2  = (const float*)d_in[4];
  const float* b2  = (const float*)d_in[5];
  const float* wf1 = (const float*)d_in[6];
  const float* bf1 = (const float*)d_in[7];
  const float* wf2 = (const float*)d_in[8];
  const float* bf2 = (const float*)d_in[9];
  float* out = (float*)d_out;

  char* ws = (char*)d_ws;
  __bf16* xwT    = (__bf16*)(ws);                        // 2 MiB  [B][32][N] bf16
  float*  h1     = (float*)(ws + ((size_t)2 << 20));     // 4 MiB  [B][N][32] fp32
  float*  pooled = (float*)(ws + ((size_t)6 << 20));     // 2 KiB  [B][32]
  __bf16* abf    = (__bf16*)(ws + ((size_t)8 << 20));    // 128 MiB bf16(A) cache
  const size_t need_abf = ((size_t)8 << 20) +
                          (size_t)B_BATCH * NNODES * NNODES * sizeof(__bf16);
  const int use_abf = (ws_size >= need_abf) ? 1 : 0;

  dim3 blk(256);
  dim3 grd(NNODES / 128, B_BATCH);

  k_transform<<<grd, blk, 0, stream>>>(x, w1, xwT, 128);
  k_aggregate<<<grd, blk, 0, stream>>>(a, abf, xwT, b1, h1, nullptr, 0, use_abf);
  k_transform<<<grd, blk, 0, stream>>>(h1, w2, xwT, 32);
  k_zero<<<1, 512, 0, stream>>>(pooled, B_BATCH * CDIM);
  k_aggregate<<<grd, blk, 0, stream>>>(a, abf, xwT, b2, nullptr, pooled, 1, use_abf);
  k_mlp<<<1, 512, 0, stream>>>(pooled, wf1, bf1, wf2, bf2, out);
}
